// OurGRU_35132832482024
// MI455X (gfx1250) — matrix operand order repaired and checked by reading, not compiled, not run on hardware
//
#include <hip/hip_runtime.h>
#include <stdint.h>

// Problem sizes (fixed by the reference)
#define B_      64
#define S_      512
#define H_      512
#define K_      1024              // H (recurrent) + 512 (input), both layers
#define NBLK    32                // persistent workgroups; each owns 16 cols/gate
#define COLS    16                // output columns per gate per block
#define THREADS 256               // 8 wave32 per block

// Dynamic LDS partition (byte offsets; dynamic LDS starts at LDS address 0)
#define HX_OFF     0
#define HX_BYTES   (B_ * K_ * 2)          // 131072 : A matrix [h|x_t] bf16, row-major
#define W_OFF      (HX_OFF + HX_BYTES)
#define W_BYTES    (3 * COLS * K_ * 2)    //  98304 : weight slice, column-major bf16
#define ZST_OFF    (W_OFF + W_BYTES)
#define ZST_BYTES  (B_ * COLS * 4)        //   4096 : z staging (f32)
#define HST_OFF    (ZST_OFF + ZST_BYTES)
#define HST_BYTES  (B_ * COLS * 4)        //   4096 : h (own cols) staging (f32)
#define SMEM_BYTES (HX_BYTES + W_BYTES + ZST_BYTES + HST_BYTES)   // 237568 < 320KB

typedef __attribute__((ext_vector_type(16))) __bf16 v16bf;
typedef __attribute__((ext_vector_type(8)))  float  v8f;

struct __attribute__((aligned(16))) U128 { unsigned int a, b, c, d; };
union Frag { v16bf v; U128 u[2]; };

__device__ __forceinline__ unsigned short f2bf(float f) {
  union { float f; unsigned int u; } x; x.f = f;
  unsigned int u = x.u + 0x7FFFu + ((x.u >> 16) & 1u);   // round-to-nearest-even
  return (unsigned short)(u >> 16);
}

__device__ __forceinline__ void grid_barrier(int* counter, int* gen) {
  __syncthreads();
  if (threadIdx.x == 0) {
    __threadfence();
    int g = atomicAdd(gen, 0);
    if (atomicAdd(counter, 1) == NBLK - 1) {
      atomicExch(counter, 0);
      __threadfence();
      atomicAdd(gen, 1);
    } else {
      while (atomicAdd(gen, 0) == g) { __builtin_amdgcn_s_sleep(2); }
    }
  }
  __syncthreads();
}

// K=1024 GEMM micro-kernel: one 16x16 output tile, software-pipelined with
// two fragment register sets so WMMA never has a WAR hazard against the
// next fragment's ds_load (kills the 4x v_nop + s_wait_dscnt 0 pattern).
__device__ __forceinline__ v8f gemm_k1024(const unsigned short* __restrict__ arow,
                                          const unsigned short* __restrict__ bcol,
                                          int half) {
#define LOAD_FRAGS(A, Bf, kk)                                    \
  do {                                                           \
    (A).u[0]  = *(const U128*)(arow + (kk) + half * 8);          \
    (A).u[1]  = *(const U128*)(arow + (kk) + 16 + half * 8);     \
    (Bf).u[0] = *(const U128*)(bcol + (kk) + half * 8);          \
    (Bf).u[1] = *(const U128*)(bcol + (kk) + 16 + half * 8);     \
  } while (0)

  Frag a0, b0, a1, b1;
  v8f acc = {0.f, 0.f, 0.f, 0.f, 0.f, 0.f, 0.f, 0.f};
  LOAD_FRAGS(a0, b0, 0);
#pragma unroll 4
  for (int kk = 0; kk < K_; kk += 64) {
    LOAD_FRAGS(a1, b1, kk + 32);
    acc = __builtin_amdgcn_wmma_f32_16x16x32_bf16(
        false, a0.v, false, b0.v, (short)0, acc, false, false);
    if (kk + 64 < K_) LOAD_FRAGS(a0, b0, kk + 64);
    acc = __builtin_amdgcn_wmma_f32_16x16x32_bf16(
        false, a1.v, false, b1.v, (short)0, acc, false, false);
  }
#undef LOAD_FRAGS
  return acc;
}

// Transpose + convert one [K_ x H_] f32 weight to column-major bf16 [H_ x K_]
__global__ void convert_weight(const float* __restrict__ src,
                               unsigned short* __restrict__ dst) {
  int idx = blockIdx.x * blockDim.x + threadIdx.x;   // over H_*K_
  if (idx >= H_ * K_) return;
  int n = idx / K_;
  int k = idx - n * K_;
  dst[idx] = f2bf(src[k * H_ + n]);
}

// Persistent GRU layer: scans all S_ timesteps with grid-wide barriers.
__global__ void __launch_bounds__(THREADS, 1)
gru_layer(const float* __restrict__ xin,
          float* __restrict__ out,
          float* __restrict__ hbuf,
          const unsigned short* __restrict__ wt,
          unsigned short* __restrict__ rhbuf,
          int* __restrict__ sync,
          float* __restrict__ hlast) {
  extern __shared__ char smem[];
  unsigned short* hx   = (unsigned short*)(smem + HX_OFF);   // [64][1024]
  unsigned short* wlds = (unsigned short*)(smem + W_OFF);    // [48][1024]
  float*          zst  = (float*)(smem + ZST_OFF);           // [64][16]
  float*          hst  = (float*)(smem + HST_OFF);           // [64][16]

  const int tid   = threadIdx.x;
  const int wv    = tid >> 5;          // wave 0..7
  const int lane  = tid & 31;
  const int half  = lane >> 4;         // 0: lanes 0-15, 1: lanes 16-31
  const int ln    = lane & 15;
  const int cbase = blockIdx.x * COLS; // this block's column base per gate

  // ---- Load this block's weight slice into LDS (once, stays for all steps) ----
  for (int i = tid * 8; i < 3 * COLS * K_; i += THREADS * 8) {
    int c = i / K_;                    // local col 0..47
    int k = i - c * K_;
    int g = c >> 4;                    // gate 0=z 1=r 2=cand
    int n = cbase + (c & 15);
    *(U128*)(wlds + c * K_ + k) =
        *(const U128*)(wt + ((size_t)(g * H_ + n)) * K_ + k);
  }
  __syncthreads();

  for (int t = 0; t < S_; ++t) {
    // ---- Phase 1: build A = [h | x_t] in LDS (bf16), stage own-col h (f32) ----
    for (int i = tid * 4; i < B_ * H_; i += THREADS * 4) {
      int row = i >> 9, col = i & 511;
      float4 f = *(const float4*)(hbuf + i);
      unsigned short* p = hx + row * K_ + col;
      p[0] = f2bf(f.x); p[1] = f2bf(f.y); p[2] = f2bf(f.z); p[3] = f2bf(f.w);
    }
    for (int i = tid * 4; i < B_ * 512; i += THREADS * 4) {
      int row = i >> 9, col = i & 511;
      float4 f = *(const float4*)(xin + (size_t)row * (S_ * 512) + t * 512 + col);
      unsigned short* p = hx + row * K_ + 512 + col;
      p[0] = f2bf(f.x); p[1] = f2bf(f.y); p[2] = f2bf(f.z); p[3] = f2bf(f.w);
    }
    for (int i = tid * 4; i < B_ * COLS; i += THREADS * 4) {
      int row = i >> 4, j = i & 15;
      *(float4*)(hst + i) = *(const float4*)(hbuf + row * H_ + cbase + j);
    }
    __syncthreads();

    // Prefetch next timestep's x slab while the GEMMs run (global_prefetch_b8)
    if (t + 1 < S_) {
      const char* nx = (const char*)(xin + (size_t)(tid >> 2) * (S_ * 512)
                                     + (t + 1) * 512 + (tid & 3) * 128);
#pragma unroll
      for (int q = 0; q < 4; ++q) __builtin_prefetch(nx + q * 512 * 4, 0, 1);
    }

    // ---- Phase 2: z/r GEMM  [64x1024] @ [1024x32] via WMMA bf16 ----
    {
      int mt   = wv >> 1;              // M tile 0..3
      int gsel = wv & 1;               // 0: z cols, 1: r cols
      const unsigned short* arow = hx + (mt * 16 + ln) * K_;
      const unsigned short* bcol = wlds + (gsel * 16 + ln) * K_;
      v8f acc = gemm_k1024(arow, bcol, half);
#pragma unroll
      for (int v = 0; v < 8; ++v) {
        int row = mt * 16 + half * 8 + v;                     // C/D layout
        float sig = 1.f / (1.f + expf(-acc[v]));
        if (gsel == 0) {
          zst[row * COLS + ln] = sig;                         // keep z for update
        } else {
          float rh = sig * hst[row * COLS + ln];              // r ⊙ h (own cols)
          rhbuf[row * H_ + cbase + ln] = f2bf(rh);            // publish slice
        }
      }
    }
    grid_barrier(sync, sync + 1);      // all r⊙h slices visible device-wide

    // ---- Phase 3: async-pull full r⊙h into A's h-half (ASYNCcnt path) ----
    for (int i = tid * 8; i < B_ * H_; i += THREADS * 8) {
      int row = i >> 9, col = i & 511;
      unsigned lds_addr = (unsigned)(HX_OFF + (row * K_ + col) * 2);
      unsigned gofs     = (unsigned)(i * 2);
      asm volatile("global_load_async_to_lds_b128 %0, %1, %2"
                   :: "v"(lds_addr), "v"(gofs), "s"(rhbuf)
                   : "memory");
    }
    asm volatile("s_wait_asynccnt 0x0" ::: "memory");
    __syncthreads();

    // ---- Candidate GEMM  [64x1024] @ [1024x16], then the GRU update ----
    if (wv < 4) {
      int mt = wv;
      const unsigned short* arow = hx + (mt * 16 + ln) * K_;
      const unsigned short* bcol = wlds + (32 + ln) * K_;     // candidate cols
      v8f acc = gemm_k1024(arow, bcol, half);
#pragma unroll
      for (int v = 0; v < 8; ++v) {
        int row = mt * 16 + half * 8 + v;
        int col = cbase + ln;
        float hh = tanhf(acc[v]);
        float z  = zst[row * COLS + ln];
        float hv = hst[row * COLS + ln];
        float hn = (1.f - z) * hv + z * hh;
        hbuf[row * H_ + col] = hn;
        out[(size_t)row * (S_ * H_) + t * H_ + col] = hn;
        if (t == S_ - 1) hlast[row * H_ + col] = hn;
      }
    }
    grid_barrier(sync, sync + 1);      // h_new visible before next step's load
  }
}

extern "C" void kernel_launch(void* const* d_in, const int* in_sizes, int n_in,
                              void* d_out, int out_size, void* d_ws, size_t ws_size,
                              hipStream_t stream) {
  (void)in_sizes; (void)n_in; (void)out_size; (void)ws_size;

  const float* x    = (const float*)d_in[0];   // [B,S,I]
  const float* hin  = (const float*)d_in[1];   // [L,B,H]
  const float* wsrc[6] = {                     // z0, r0, c0, z1, r1, c1
      (const float*)d_in[2], (const float*)d_in[3], (const float*)d_in[4],
      (const float*)d_in[5], (const float*)d_in[6], (const float*)d_in[7]};

  float* out1  = (float*)d_out;                        // [B,S,H]
  float* hlast = (float*)d_out + (size_t)B_ * S_ * H_; // [L,B,H]

  // Workspace layout
  char* ws = (char*)d_ws;
  int* sync = (int*)ws;                                           // 2 pairs
  size_t off = 256;
  float* hbuf = (float*)(ws + off);          off += (size_t)2 * B_ * H_ * 4;
  unsigned short* rhbuf = (unsigned short*)(ws + off); off += (size_t)B_ * H_ * 2;
  unsigned short* wt = (unsigned short*)(ws + off);    off += (size_t)6 * H_ * K_ * 2;
  float* out0 = (float*)(ws + off);          off += (size_t)B_ * S_ * H_ * 4;

  hipFuncSetAttribute((const void*)gru_layer,
                      hipFuncAttributeMaxDynamicSharedMemorySize, SMEM_BYTES);

  hipMemsetAsync(sync, 0, 256, stream);
  hipMemcpyAsync(hbuf, hin, (size_t)2 * B_ * H_ * sizeof(float),
                 hipMemcpyDeviceToDevice, stream);

  const int cvt_blocks = (H_ * K_ + 255) / 256;
  for (int m = 0; m < 6; ++m)
    convert_weight<<<cvt_blocks, 256, 0, stream>>>(
        wsrc[m], wt + (size_t)m * H_ * K_);

  // Layer 0: input x -> out0
  gru_layer<<<NBLK, THREADS, SMEM_BYTES, stream>>>(
      x, out0, hbuf, wt, rhbuf, sync, hlast);
  // Layer 1: input out0 -> out1
  gru_layer<<<NBLK, THREADS, SMEM_BYTES, stream>>>(
      out0, out1, hbuf + (size_t)B_ * H_, wt + (size_t)3 * H_ * K_,
      rhbuf, sync + 4, hlast + (size_t)B_ * H_);
}
